// Aggregator_76501957477036
// MI455X (gfx1250) — matrix operand order repaired and checked by reading, not compiled, not run on hardware
//
#include <hip/hip_runtime.h>

// ---------------------------------------------------------------------------
// Fused Aggregator for MI455X (gfx1250, wave32, WMMA bf16).
//   gates = sigmoid(nodes @ Wg.T + bg)
//   data  =          nodes @ Wt.T + bt
//   res   = fmask @ (data * gates)            // [256 x 256] f32
//
// Design: persistent split-K blocks; each block loops over 128-node chunks:
//   stage nodes f32->bf16 into LDS (A-fragment swizzle) -> phase1 WMMA GEMMs
//   vs packed weights + sigmoid*mul -> Y bf16 in LDS (B-fragment swizzle) ->
//   phase2 WMMA mask-matmul accumulating [256x256] f32 in VGPRs.
// Partials written to d_ws, reduced by a final kernel (deterministic).
// ---------------------------------------------------------------------------

#define N_NODES 200000
#define D       256      // D_IN == D_OUT
#define BATCH   256
#define CHUNK   128
#define NWAVES  16
#define TPB     (NWAVES * 32)
#define OUT_ELEMS (BATCH * D)

static_assert(N_NODES % 32 == 0, "n-blocks must tile N exactly");

typedef __attribute__((ext_vector_type(16))) __bf16 v16bf;
typedef __attribute__((ext_vector_type(8)))  float  v8f;

union Frag {
    v16bf v;
    unsigned short u[16];
};

__device__ __forceinline__ unsigned short f2bf(float f) {
    unsigned int u = __float_as_uint(f);
    u += 0x7FFFu + ((u >> 16) & 1u);          // round-to-nearest-even
    return (unsigned short)(u >> 16);
}

__device__ __forceinline__ v8f v8f_zero() {
    v8f r;
    #pragma unroll
    for (int i = 0; i < 8; ++i) r[i] = 0.0f;
    return r;
}

// ---------------------------------------------------------------------------
// Kernel 0: pack Wt and Wg (f32, [d][k] row-major) into bf16 WMMA B-fragment
// layout: frag index fi = (kb*16 + db)*32 + lane holds 16 contiguous bf16:
//   lane<16 : column d = db*16+lane,      k = kb*32 + (j&7) + 16*(j>>3)      (j=0..15)
//   lane>=16: column d = db*16+(lane-16), k = kb*32 + 8 + (j&7) + 16*(j>>3)
// ---------------------------------------------------------------------------
__global__ void pack_weights(const float* __restrict__ Wt,
                             const float* __restrict__ Wg,
                             unsigned short* __restrict__ wpack) {
    int idx = blockIdx.x * blockDim.x + threadIdx.x;   // 0..131071
    const float* W = (idx < 65536) ? Wt : Wg;
    int p    = idx & 65535;
    int j    = p & 15;
    int lane = (p >> 4) & 31;
    int db   = (p >> 9) & 15;
    int kb   = p >> 13;
    int k = kb * 32 + ((lane >> 4) & 1) * 8 + (j & 7) + (j >> 3) * 16;
    int d = db * 16 + (lane & 15);
    wpack[idx] = f2bf(W[d * D + k]);
}

// ---------------------------------------------------------------------------
// Fused kernel: 16 waves / block.
// Phase 1 roles : wave w -> m-tile (w&7), d-strips ((w>>3)*8 .. +7)
// Phase 2 roles : wave w -> b-tile w (16 batch rows), all 16 d-strips
// ---------------------------------------------------------------------------
__global__ __launch_bounds__(TPB, 1)
void fused_agg(const float* __restrict__ nodes,
               const int*   __restrict__ masks,
               const unsigned short* __restrict__ wtp,
               const unsigned short* __restrict__ wgp,
               const float* __restrict__ bt,
               const float* __restrict__ bg,
               float* __restrict__ partials) {
    // 64 KB nodes tile (bf16, A-fragment swizzled: [mt(8)][kb(8)][lane(32)][16])
    __shared__ __align__(16) unsigned short ldsN[8 * 8 * 32 * 16];
    // 64 KB Y tile (bf16, B-fragment swizzled: [nb(4)][db(16)][lane(32)][16])
    __shared__ __align__(16) unsigned short ldsY[4 * 16 * 32 * 16];

    const int tid  = threadIdx.x;
    const int w    = tid >> 5;
    const int lane = tid & 31;

    v8f acc[16];                               // [256 x 256] f32 across 16 waves
    #pragma unroll
    for (int i = 0; i < 16; ++i) acc[i] = v8f_zero();

    const int numChunks = (N_NODES + CHUNK - 1) / CHUNK;
    const int mt   = w & 7;
    const int dgrp = (w >> 3) * 8;

    for (int c = blockIdx.x; c < numChunks; c += gridDim.x) {
        const int start   = c * CHUNK;
        const int nvalid  = min(CHUNK, N_NODES - start);
        const int mtiles  = nvalid >> 4;
        const int nblocks = nvalid >> 5;

        // ---- stage nodes chunk: f32 -> bf16 into A-fragment layout ----
        const int nquads = nvalid * (D / 4);
        for (int i = tid; i < nquads; i += TPB) {
            const int r  = i >> 6;
            const int q  = i & 63;
            const int k0 = q * 4;
            const float4 f =
                ((const float4*)(nodes + (size_t)(start + r) * D))[q];
            const unsigned int lo = (unsigned)f2bf(f.x) | ((unsigned)f2bf(f.y) << 16);
            const unsigned int hi = (unsigned)f2bf(f.z) | ((unsigned)f2bf(f.w) << 16);
            const int kk    = k0 & 31;
            const int kb    = k0 >> 5;
            const int half  = (kk >> 3) & 1;
            const int slane = (r & 15) + 16 * half;
            const int j0    = (kk & 7) + 8 * ((kk >> 4) & 1);
            const int idx   = (((r >> 4) * 8 + kb) * 32 + slane) * 16 + j0;
            *(uint2*)(&ldsN[idx]) = make_uint2(lo, hi);
        }
        __syncthreads();

        // ---- prefetch next chunk of nodes into caches ----
        {
            const int nc = c + gridDim.x;
            if (nc < numChunks) {
                const int ns  = nc * CHUNK;
                const int nvn = min(CHUNK, N_NODES - ns);
                #pragma unroll
                for (int q = 0; q < 2; ++q) {
                    const int elem = (tid * 2 + q) * 32;   // one 128B line each
                    if (elem < nvn * D)
                        __builtin_prefetch(nodes + (size_t)ns * D + elem, 0, 1);
                }
            }
        }

        // ---- phase 1: Y[mt, d-strip] = (A@Wt + bt) * sigmoid(A@Wg + bg) ----
        if (mt < mtiles) {
            Frag a[8];                          // cache A fragments (K = 256)
            #pragma unroll
            for (int kb = 0; kb < 8; ++kb)
                a[kb].v = *(const v16bf*)(&ldsN[((mt * 8 + kb) * 32 + lane) * 16]);

            for (int d2 = 0; d2 < 8; ++d2) {
                const int db   = dgrp + d2;
                const int dcol = db * 16 + (lane & 15);
                const float btv = bt[dcol];
                const float bgv = bg[dcol];
                v8f at = v8f_zero();
                v8f ag = v8f_zero();
                #pragma unroll
                for (int kb = 0; kb < 8; ++kb) {
                    const int fi = (kb * 16 + db) * 32 + lane;
                    const v16bf bw = ((const v16bf*)wtp)[fi];
                    const v16bf gw = ((const v16bf*)wgp)[fi];
                    at = __builtin_amdgcn_wmma_f32_16x16x32_bf16(
                             false, a[kb].v, false, bw, (short)0, at, false, false);
                    ag = __builtin_amdgcn_wmma_f32_16x16x32_bf16(
                             false, a[kb].v, false, gw, (short)0, ag, false, false);
                }
                // epilogue: bias + sigmoid gate, pack 8 bf16, one 16B DS store.
                unsigned short ys[8];
                #pragma unroll
                for (int v = 0; v < 8; ++v) {
                    const float t = at[v] + btv;
                    float g = ag[v] + bgv;
                    g = 1.0f / (1.0f + __expf(-g));
                    ys[v] = f2bf(t * g);
                }
                uint4 qv;
                qv.x = (unsigned)ys[0] | ((unsigned)ys[1] << 16);
                qv.y = (unsigned)ys[2] | ((unsigned)ys[3] << 16);
                qv.z = (unsigned)ys[4] | ((unsigned)ys[5] << 16);
                qv.w = (unsigned)ys[6] | ((unsigned)ys[7] << 16);
                // C-tile (m,d) -> B-fragment slot; collapses to lane-local 16B
                const int base =
                    (((mt >> 1) * 16 + db) * 32 + lane) * 16 + 8 * (mt & 1);
                *(uint4*)(&ldsY[base]) = qv;
            }
        }
        __syncthreads();

        // ---- phase 2: acc[b-tile w] += fmask[b, n-block] @ Y[n-block, :] ----
        for (int nb = 0; nb < nblocks; ++nb) {
            const int nbase = start + nb * 32;
            const int brow  = w * 16 + (lane & 15);
            const int koff  = (lane < 16) ? 0 : 8;
            const int* mrow = masks + (size_t)brow * N_NODES + nbase + koff;
            union { int4 a4[4]; int s[16]; } mraw;
            mraw.a4[0] = *(const int4*)(mrow);
            mraw.a4[1] = *(const int4*)(mrow + 4);
            mraw.a4[2] = *(const int4*)(mrow + 16);
            mraw.a4[3] = *(const int4*)(mrow + 20);
            Frag mf;                            // 0/1 int -> bf16 exactly
            #pragma unroll
            for (int j = 0; j < 16; ++j)
                mf.u[j] = mraw.s[j] ? 0x3F80u : 0u;

            #pragma unroll
            for (int db = 0; db < 16; ++db) {
                const v16bf yf =
                    *(const v16bf*)(&ldsY[((nb * 16 + db) * 32 + lane) * 16]);
                acc[db] = __builtin_amdgcn_wmma_f32_16x16x32_bf16(
                              false, mf.v, false, yf, (short)0, acc[db],
                              false, false);
            }
        }
        __syncthreads();   // protect ldsY/ldsN before next iteration rewrites
    }

    // ---- write per-block partial [256 x 256] ----
    float* out = partials + (size_t)blockIdx.x * OUT_ELEMS;
    #pragma unroll
    for (int db = 0; db < 16; ++db) {
        #pragma unroll
        for (int v = 0; v < 8; ++v) {
            const int b = w * 16 + v + 8 * (lane >> 4);
            const int d = db * 16 + (lane & 15);
            out[b * D + d] = acc[db][v];
        }
    }
}

// ---------------------------------------------------------------------------
// Split-K reduction: out = sum_p partials[p]
// ---------------------------------------------------------------------------
__global__ void reduce_partials(const float* __restrict__ partials,
                                float* __restrict__ out, int P) {
    const int idx = blockIdx.x * blockDim.x + threadIdx.x;
    if (idx >= OUT_ELEMS) return;
    float s = 0.0f;
    for (int p = 0; p < P; ++p)
        s += partials[(size_t)p * OUT_ELEMS + idx];
    out[idx] = s;
}

// ---------------------------------------------------------------------------
extern "C" void kernel_launch(void* const* d_in, const int* in_sizes, int n_in,
                              void* d_out, int out_size, void* d_ws, size_t ws_size,
                              hipStream_t stream) {
    const float* nodes = (const float*)d_in[0];
    const int*   masks = (const int*)  d_in[1];
    const float* Wt    = (const float*)d_in[2];
    const float* bt    = (const float*)d_in[3];
    const float* Wg    = (const float*)d_in[4];
    const float* bg    = (const float*)d_in[5];

    // workspace layout: [wt_pack bf16 128KB][wg_pack bf16 128KB][partials f32]
    unsigned short* wtp = (unsigned short*)d_ws;
    unsigned short* wgp = wtp + 65536;
    float* partials = (float*)((char*)d_ws + 262144);

    size_t avail = (ws_size > 262144)
                       ? (ws_size - 262144) / (sizeof(float) * OUT_ELEMS)
                       : 1;
    int P = (int)(avail < 1 ? 1 : (avail > 256 ? 256 : avail));

    pack_weights<<<512, 256, 0, stream>>>(Wt, Wg, wtp);
    fused_agg<<<P, TPB, 0, stream>>>(nodes, masks, wtp, wgp, bt, bg, partials);
    reduce_partials<<<(OUT_ELEMS + 255) / 256, 256, 0, stream>>>(
        partials, (float*)d_out, P);
}
